// StageAttentionwCAM_7327214207692
// MI455X (gfx1250) — matrix operand order repaired and checked by reading, not compiled, not run on hardware
//
#include <hip/hip_runtime.h>

// ---------------------------------------------------------------------------
// MI455X (gfx1250) fused DANet dual-attention pipeline.
// Memory-bound (~0.5 GB traffic @ 23.3 TB/s ~ 20 us; ~8 GFLOP negligible on
// the matrix pipe). GEMMs -> v_wmma_f32_16x16x32_f16 (fp32 accumulate).
// Conv1x1 X strips are staged into LDS by the Tensor Data Mover
// (tensor_load_to_lds + s_wait_tensorcnt), weights staged as f16 in LDS.
// ---------------------------------------------------------------------------

typedef __attribute__((ext_vector_type(16))) _Float16 v16h;
typedef __attribute__((ext_vector_type(8)))  float    v8f;
typedef unsigned int __attribute__((ext_vector_type(4))) u32x4;
typedef int          __attribute__((ext_vector_type(4))) i32x4;
typedef int          __attribute__((ext_vector_type(8))) i32x8;

#define BB 4
#define CC 64
#define C8V 8
#define HH 128
#define WWID 128
#define HW 16384            // HH*WWID
#define PLANE (BB * CC * HW)

// --- WMMA 16x16x32 f16 per-lane element index helpers (ISA 7.12.2, wave32) ---
// A (16x32, f16): lane L holds row m=L%16; element h of v16h maps to
//   k = (h/8)*16 + (L/16)*8 + (h%8)   (8 consecutive halfs -> ds_load_b128)
__device__ __forceinline__ int a_kidx(int lane, int h) {
  return ((h >> 3) << 4) + ((lane >> 4) << 3) + (h & 7);
}
// B (32x16, f16): lane L holds col n=L%16; element h maps to
//   k = (h/8)*16 + (L/16)*8 + (h%8)
__device__ __forceinline__ int b_kidx(int lane, int h) {
  return ((h >> 3) << 4) + ((lane >> 4) << 3) + (h & 7);
}
// C/D (16x16, f32): lane L, vgpr r -> m = r + (L/16)*8, n = L%16

// ---------------------------------------------------------------------------
// 2x2 max-pool (B,C,256,256) -> (B,C,128,128)
// ---------------------------------------------------------------------------
__global__ void maxpool2_kernel(const float* __restrict__ fb, float* __restrict__ fbd) {
  long i = (long)blockIdx.x * blockDim.x + threadIdx.x;
  if (i >= (long)PLANE) return;
  int w = (int)(i & 127);
  int h = (int)((i >> 7) & 127);
  long bc = i >> 14;
  const float* p = fb + (bc * 256 + 2 * h) * 256 + 2 * w;
  fbd[i] = fmaxf(fmaxf(p[0], p[1]), fmaxf(p[256], p[257]));
}

// ---------------------------------------------------------------------------
// 1x1 conv as GEMM: O[b,m,p] = sum_c W[m,c] * X[b,c,p] + bias[m].  C == 64.
// One block = one (b, 128-pixel strip). X strip (64ch x 128px f32, 32 KB)
// staged into LDS by the Tensor Data Mover; weights staged as f16.
// Each of 8 waves computes a 16-pixel tile across all M tiles.
// wstride: per-batch weight stride (0 = shared weights, 4096 = CAM attention)
// ---------------------------------------------------------------------------
template <int COUT>
__global__ void conv1x1_wmma_kernel(const float* __restrict__ X,
                                    const float* __restrict__ Wm,
                                    const float* __restrict__ bias,
                                    float* __restrict__ O, long wstride) {
  constexpr int NMT = (COUT + 15) / 16;
  const int stripsPerBatch = HW / 128;                 // 128
  int b  = blockIdx.x / stripsPerBatch;
  int p0 = (blockIdx.x % stripsPerBatch) * 128;
  const float* Xb = X + (long)b * CC * HW + p0;        // tile start (c=0, p=p0)
  const float* Wb = Wm + (long)b * wstride;

  __shared__ float    xs[CC * 128];                    // 32 KB  [c][p]
  __shared__ _Float16 wsh[NMT * 16 * CC];              // <=8 KB [m][c] f16

#if defined(__gfx1250__) && __has_builtin(__builtin_amdgcn_tensor_load_to_lds)
  if (threadIdx.x < 32) {                              // wave 0 issues the DMA
    // Tensor DMA descriptor (ISA 8.3-8.5): 2D tile, dim0 = 128 contiguous
    // pixels, dim1 = 64 channels, stride tensor_dim0_stride = HW, f32.
    unsigned long long ga = (unsigned long long)(uintptr_t)Xb;
    unsigned int lds = (unsigned int)(uintptr_t)(void*)xs;  // addr[31:0] = LDS offset
    u32x4 g0 = { 1u,                                   // [1:0] count=1
                 lds,                                  // [63:32] lds_addr
                 (unsigned int)ga,                     // [95:64] global_addr lo
                 (unsigned int)(ga >> 32) | (2u << 30) }; // ga hi | type=2
    i32x8 g1 = { (int)(2u << 16),                      // [17:16] data_size=4B
                 (int)(((unsigned)HW & 0xFFFFu) << 16),// [79:48] tensor_dim0 lo16
                 (int)(((unsigned)HW >> 16) | ((unsigned)CC << 16)), // dim0 hi | tensor_dim1 lo
                 (int)(128u << 16),                    // [127:112] tile_dim0 = 128
                 (int)CC,                              // [143:128] tile_dim1 = 64
                 (int)HW,                              // [207:160] tensor_dim0_stride lo32
                 0, 0 };
    i32x4 gz = { 0, 0, 0, 0 };
#if __clang_major__ >= 23
    i32x8 gz8 = { 0, 0, 0, 0, 0, 0, 0, 0 };
    __builtin_amdgcn_tensor_load_to_lds(g0, g1, gz, gz, gz8, 0);
#else
    __builtin_amdgcn_tensor_load_to_lds(g0, g1, gz, gz, 0);
#endif
    __builtin_amdgcn_s_wait_tensorcnt(0);              // DMA complete in LDS
  }
#else
  for (int i = threadIdx.x; i < CC * 128; i += blockDim.x)
    xs[i] = Xb[((long)(i >> 7)) * HW + (i & 127)];
#endif

  for (int i = threadIdx.x; i < NMT * 16 * CC; i += blockDim.x) {
    int m = i / CC, c = i % CC;
    wsh[i] = (m < COUT) ? (_Float16)Wb[m * CC + c] : (_Float16)0.0f;
  }
  __syncthreads();

  int wv = threadIdx.x >> 5, lane = threadIdx.x & 31;
  int n = lane & 15;
  int pl = wv * 16 + n;                                // local pixel column

  v8f acc[NMT];
#pragma unroll
  for (int i = 0; i < NMT; ++i) acc[i] = {};

#pragma unroll
  for (int ks = 0; ks < 2; ++ks) {                     // K = 64 in 2 steps of 32
    int c0 = ks * 32;
    v16h bt;
#pragma unroll
    for (int h = 0; h < 16; ++h)
      bt[h] = (_Float16)xs[(c0 + b_kidx(lane, h)) * 128 + pl];
#pragma unroll
    for (int mt = 0; mt < NMT; ++mt) {
      v16h at;
#pragma unroll
      for (int h = 0; h < 16; ++h)
        at[h] = wsh[(mt * 16 + (lane & 15)) * CC + c0 + a_kidx(lane, h)];
      acc[mt] = __builtin_amdgcn_wmma_f32_16x16x32_f16(
          false, at, false, bt, (short)0, acc[mt], false, false);
    }
  }

  float* Ob = O + (long)b * COUT * HW;
#pragma unroll
  for (int mt = 0; mt < NMT; ++mt) {
#pragma unroll
    for (int r = 0; r < 8; ++r) {
      int m = mt * 16 + r + ((lane >> 4) << 3);
      if (m < COUT) {
        float bv = bias ? bias[m] : 0.0f;
        Ob[(long)m * HW + p0 + wv * 16 + n] = acc[mt][r] + bv;
      }
    }
  }
}

// ---------------------------------------------------------------------------
// BatchNorm batch statistics: per-channel scale/shift from y (pre-BN)
// ---------------------------------------------------------------------------
__global__ void bn_stats_kernel(const float* __restrict__ y,
                                const float* __restrict__ bn_g,
                                const float* __restrict__ bn_b,
                                float* __restrict__ scale,
                                float* __restrict__ shift) {
  int c = blockIdx.x;
  __shared__ float ssum[256], ssq[256];
  float s = 0.f, q = 0.f;
  for (int b = 0; b < BB; ++b) {
    const float* p = y + (long)(b * CC + c) * HW;
    for (int i = threadIdx.x; i < HW; i += 256) {
      float v = p[i];
      s += v; q += v * v;
    }
  }
  ssum[threadIdx.x] = s; ssq[threadIdx.x] = q;
  __syncthreads();
  for (int st = 128; st > 0; st >>= 1) {
    if ((int)threadIdx.x < st) {
      ssum[threadIdx.x] += ssum[threadIdx.x + st];
      ssq[threadIdx.x]  += ssq[threadIdx.x + st];
    }
    __syncthreads();
  }
  if (threadIdx.x == 0) {
    float n = (float)(BB * HW);
    float mu = ssum[0] / n;
    float var = ssq[0] / n - mu * mu;
    float sc = bn_g[c] * rsqrtf(var + 1e-5f);
    scale[c] = sc;
    shift[c] = bn_b[c] - mu * sc;
  }
}

__global__ void bn_relu_kernel(float* __restrict__ y,
                               const float* __restrict__ scale,
                               const float* __restrict__ shift) {
  long i = (long)blockIdx.x * blockDim.x + threadIdx.x;
  if (i >= (long)PLANE) return;
  int c = (int)((i >> 14) & 63);
  y[i] = fmaxf(0.0f, y[i] * scale[c] + shift[c]);
}

// ---------------------------------------------------------------------------
// Criss-cross logits + joint softmax over [H column | W row] = 256 logits.
// One block per (b,w); thread h owns pixel (h,w). K column staged in LDS.
// Raw logits written to ah/aw, then normalized in place (online softmax).
// ---------------------------------------------------------------------------
__global__ void cc_softmax_kernel(const float* __restrict__ q,
                                  const float* __restrict__ kk,
                                  float* __restrict__ ah,
                                  float* __restrict__ aw) {
  int b = blockIdx.x / WWID;
  int w = blockIdx.x % WWID;
  int h = threadIdx.x;           // 0..127
  __shared__ float kcol[C8V * HH];
  for (int i = threadIdx.x; i < C8V * HH; i += blockDim.x) {
    int c = i >> 7, g = i & 127;
    kcol[i] = kk[((long)(b * C8V + c)) * HW + g * WWID + w];
  }
  __syncthreads();

  float qv[C8V];
#pragma unroll
  for (int c = 0; c < C8V; ++c)
    qv[c] = q[((long)(b * C8V + c)) * HW + h * WWID + w];

  float m = -3.0e38f, s = 0.f;
  float* ahp = ah + (((long)(b * HH + h)) * WWID + w) * HH;
  for (int g = 0; g < HH; ++g) {
    float e = 0.f;
#pragma unroll
    for (int c = 0; c < C8V; ++c) e += qv[c] * kcol[c * HH + g];
    if (g == h) e += -1.0e30f;   // diagonal mask (matches reference +NEG_INF)
    float nm = fmaxf(m, e);
    s = s * __expf(m - nm) + __expf(e - nm);
    m = nm;
    ahp[g] = e;
  }
  float* awp = aw + (((long)(b * HH + h)) * WWID + w) * WWID;
  const float* krow = kk + (long)b * C8V * HW + h * WWID;   // [c*HW + vv]
  for (int vv = 0; vv < WWID; ++vv) {
    __builtin_prefetch(krow + vv + 16, 0, 1);               // global_prefetch_b8
    float e = 0.f;
#pragma unroll
    for (int c = 0; c < C8V; ++c) e += qv[c] * krow[(long)c * HW + vv];
    float nm = fmaxf(m, e);
    s = s * __expf(m - nm) + __expf(e - nm);
    m = nm;
    awp[vv] = e;
  }
  float inv = 1.0f / s;
  for (int g = 0; g < HH; ++g)  ahp[g] = __expf(ahp[g] - m) * inv;
  for (int vv = 0; vv < WWID; ++vv) awp[vv] = __expf(awp[vv] - m) * inv;
}

// ---------------------------------------------------------------------------
// out_h[c,h] (fixed b,w) = sum_g v[b,c,g,w] * ah[b,h,w,g]  — 64x128x128 GEMM.
// V column + attention matrix staged in LDS as f16, WMMA from ds gathers.
// ---------------------------------------------------------------------------
__global__ void cc_out_h_kernel(const float* __restrict__ v,
                                const float* __restrict__ ah,
                                float* __restrict__ oh) {
  int b = blockIdx.x / WWID;
  int w = blockIdx.x % WWID;
  __shared__ _Float16 vcs[CC * HH];    // [c][g] = v[b,c,g,w]
  __shared__ _Float16 ahs[HH * HH];    // [h][g]
  const float* vbase = v + (long)b * CC * HW + w;
  for (int i = threadIdx.x; i < CC * HH; i += blockDim.x) {
    int c = i >> 7, g = i & 127;
    vcs[i] = (_Float16)vbase[((long)c * HH + g) * WWID];
  }
  for (int i = threadIdx.x; i < HH * HH; i += blockDim.x) {
    int h = i >> 7, g = i & 127;
    ahs[i] = (_Float16)ah[(((long)(b * HH + h)) * WWID + w) * HH + g];
  }
  __syncthreads();

  int wv = threadIdx.x >> 5, lane = threadIdx.x & 31;
  int n = lane & 15;
  for (int t = wv; t < 32; t += 8) {     // 4 M-tiles x 8 N-tiles
    int mt = t & 3, nt = t >> 2;
    v8f acc = {};
#pragma unroll
    for (int ks = 0; ks < 4; ++ks) {     // K = 128
      int g0 = ks * 32;
      v16h at, bt;
#pragma unroll
      for (int h = 0; h < 16; ++h) {
        at[h] = vcs[(mt * 16 + (lane & 15)) * HH + g0 + a_kidx(lane, h)];
        bt[h] = ahs[(nt * 16 + n) * HH + g0 + b_kidx(lane, h)];
      }
      acc = __builtin_amdgcn_wmma_f32_16x16x32_f16(
          false, at, false, bt, (short)0, acc, false, false);
    }
#pragma unroll
    for (int r = 0; r < 8; ++r) {
      int c = mt * 16 + r + ((lane >> 4) << 3);
      int h = nt * 16 + n;
      oh[((long)(b * CC + c) * HH + h) * WWID + w] = acc[r];
    }
  }
}

// ---------------------------------------------------------------------------
// out_w[c,w] (fixed b,h) = sum_v v[b,c,h,vv] * aw[b,h,w,vv], then
// sout = gamma*(oh + out_w) + base   (base = x or y)
// ---------------------------------------------------------------------------
__global__ void cc_out_w_combine_kernel(const float* __restrict__ v,
                                        const float* __restrict__ aw,
                                        const float* __restrict__ oh,
                                        const float* __restrict__ base,
                                        const float* __restrict__ gamma,
                                        float* __restrict__ sout) {
  int b = blockIdx.x / HH;
  int h = blockIdx.x % HH;
  __shared__ _Float16 vrs[CC * WWID];    // [c][vv] = v[b,c,h,vv] (contiguous)
  __shared__ _Float16 aws[WWID * WWID];  // [w][vv] (contiguous)
  for (int i = threadIdx.x; i < CC * WWID; i += blockDim.x) {
    int c = i >> 7, vv = i & 127;
    vrs[i] = (_Float16)v[((long)(b * CC + c) * HH + h) * WWID + vv];
  }
  for (int i = threadIdx.x; i < WWID * WWID; i += blockDim.x) {
    int w = i >> 7, vv = i & 127;
    aws[i] = (_Float16)aw[(((long)(b * HH + h)) * WWID + w) * WWID + vv];
  }
  __syncthreads();

  float g = gamma[0];
  int wv = threadIdx.x >> 5, lane = threadIdx.x & 31;
  int n = lane & 15;
  for (int t = wv; t < 32; t += 8) {
    int mt = t & 3, nt = t >> 2;
    v8f acc = {};
#pragma unroll
    for (int ks = 0; ks < 4; ++ks) {
      int k0 = ks * 32;
      v16h at, bt;
#pragma unroll
      for (int hh = 0; hh < 16; ++hh) {
        at[hh] = vrs[(mt * 16 + (lane & 15)) * WWID + k0 + a_kidx(lane, hh)];
        bt[hh] = aws[(nt * 16 + n) * WWID + k0 + b_kidx(lane, hh)];
      }
      acc = __builtin_amdgcn_wmma_f32_16x16x32_f16(
          false, at, false, bt, (short)0, acc, false, false);
    }
#pragma unroll
    for (int r = 0; r < 8; ++r) {
      int c = mt * 16 + r + ((lane >> 4) << 3);
      int w = nt * 16 + n;
      long idx = ((long)(b * CC + c) * HH + h) * WWID + w;
      sout[idx] = g * (oh[idx] + acc[r]) + base[idx];
    }
  }
}

// ---------------------------------------------------------------------------
// CAM Gram matrix: energy[b,c,d] = sum_n f[b,c,n]*f[b,d,n], K = 16384 (WMMA)
// ---------------------------------------------------------------------------
__global__ void cam_energy_kernel(const float* __restrict__ f,
                                  float* __restrict__ energy) {
  int b = blockIdx.x;
  int wv = threadIdx.x >> 5, lane = threadIdx.x & 31;
  const float* fbb = f + (long)b * CC * HW;
  int n = lane & 15;
  for (int t = wv; t < 16; t += 8) {     // 4x4 (c,d) tiles, 2 per wave
    int ct = t & 3, dt = t >> 2;
    v8f acc = {};
    for (int n0 = 0; n0 < HW; n0 += 32) {
      v16h at, bt;
#pragma unroll
      for (int hh = 0; hh < 16; ++hh) {
        at[hh] = (_Float16)fbb[(long)(ct * 16 + (lane & 15)) * HW + n0 + a_kidx(lane, hh)];
        bt[hh] = (_Float16)fbb[(long)(dt * 16 + n) * HW + n0 + b_kidx(lane, hh)];
      }
      acc = __builtin_amdgcn_wmma_f32_16x16x32_f16(
          false, at, false, bt, (short)0, acc, false, false);
    }
#pragma unroll
    for (int r = 0; r < 8; ++r) {
      int c = ct * 16 + r + ((lane >> 4) << 3);
      energy[(long)b * 4096 + c * 64 + dt * 16 + n] = acc[r];
    }
  }
}

// softmax of (rowmax - energy) along d; one thread per channel row
__global__ void cam_att_kernel(const float* __restrict__ energy,
                               float* __restrict__ att) {
  int b = blockIdx.x;
  int c = threadIdx.x;          // 0..63
  const float* e = energy + (long)b * 4096 + c * 64;
  float mx = -3.0e38f, mn = 3.0e38f;
  for (int d = 0; d < 64; ++d) {
    float v = e[d];
    mx = fmaxf(mx, v);
    mn = fminf(mn, v);
  }
  float m2 = mx - mn;           // max of energy_new = rowmax - min(energy)
  float s = 0.f;
  for (int d = 0; d < 64; ++d) s += __expf((mx - e[d]) - m2);
  float inv = 1.0f / s;
  float* a = att + (long)b * 4096 + c * 64;
  for (int d = 0; d < 64; ++d) a[d] = __expf((mx - e[d]) - m2) * inv;
}

// ---------------------------------------------------------------------------
// Final fuse: out = 2*x + pam_g*sx + pam_g2*sf + cam_g*camv
// ---------------------------------------------------------------------------
__global__ void final_combine_kernel(const float* __restrict__ x,
                                     const float* __restrict__ sx,
                                     const float* __restrict__ sf,
                                     const float* __restrict__ camv,
                                     const float* __restrict__ pg,
                                     const float* __restrict__ pg2,
                                     const float* __restrict__ cg,
                                     float* __restrict__ out) {
  long i = (long)blockIdx.x * blockDim.x + threadIdx.x;
  if (i >= (long)PLANE) return;
  out[i] = 2.0f * x[i] + pg[0] * sx[i] + pg2[0] * sf[i] + cg[0] * camv[i];
}

// ---------------------------------------------------------------------------
extern "C" void kernel_launch(void* const* d_in, const int* in_sizes, int n_in,
                              void* d_out, int out_size, void* d_ws, size_t ws_size,
                              hipStream_t stream) {
  const float* x       = (const float*)d_in[0];
  const float* fbin    = (const float*)d_in[1];
  const float* w_conv  = (const float*)d_in[2];
  const float* b_conv  = (const float*)d_in[3];
  const float* bn_g    = (const float*)d_in[4];
  const float* bn_b    = (const float*)d_in[5];
  const float* xq_w    = (const float*)d_in[6];
  const float* xq_b    = (const float*)d_in[7];
  const float* xk_w    = (const float*)d_in[8];
  const float* xk_b    = (const float*)d_in[9];
  const float* xv_w    = (const float*)d_in[10];
  const float* xv_b    = (const float*)d_in[11];
  const float* x_gamma = (const float*)d_in[12];
  const float* fq_w    = (const float*)d_in[13];
  const float* fq_b    = (const float*)d_in[14];
  const float* fk_w    = (const float*)d_in[15];
  const float* fk_b    = (const float*)d_in[16];
  const float* fv_w    = (const float*)d_in[17];
  const float* fv_b    = (const float*)d_in[18];
  const float* f_gamma = (const float*)d_in[19];
  const float* pam_g   = (const float*)d_in[20];
  const float* pam_g2  = (const float*)d_in[21];
  const float* cam_g   = (const float*)d_in[22];
  float* out = (float*)d_out;

  float* ws = (float*)d_ws;
  size_t o = 0;
  float* fbd    = ws + o; o += (size_t)PLANE;
  float* y      = ws + o; o += (size_t)PLANE;
  float* qb     = ws + o; o += (size_t)BB * C8V * HW;
  float* kb     = ws + o; o += (size_t)BB * C8V * HW;
  float* vb     = ws + o; o += (size_t)PLANE;
  float* oh     = ws + o; o += (size_t)PLANE;
  float* sx     = ws + o; o += (size_t)PLANE;
  float* sf     = ws + o; o += (size_t)PLANE;
  float* camv   = ws + o; o += (size_t)PLANE;
  float* ah     = ws + o; o += (size_t)BB * HH * WWID * HH;
  float* aw     = ws + o; o += (size_t)BB * HH * WWID * WWID;
  float* energy = ws + o; o += (size_t)BB * 64 * 64;
  float* attb   = ws + o; o += (size_t)BB * 64 * 64;
  float* scale  = ws + o; o += 64;
  float* shift  = ws + o; o += 64;

  const int ELEM_BLK = (PLANE + 255) / 256;        // 16384
  const int CONV_BLK = BB * (HW / 128);            // 512 blocks, 8 waves each

  // ---- y branch: maxpool -> conv -> BN(batch stats) -> relu ----
  maxpool2_kernel<<<ELEM_BLK, 256, 0, stream>>>(fbin, fbd);
  conv1x1_wmma_kernel<64><<<CONV_BLK, 256, 0, stream>>>(fbd, w_conv, b_conv, y, 0);
  bn_stats_kernel<<<64, 256, 0, stream>>>(y, bn_g, bn_b, scale, shift);
  bn_relu_kernel<<<ELEM_BLK, 256, 0, stream>>>(y, scale, shift);

  // ---- s_module on x -> sx ----
  conv1x1_wmma_kernel<8><<<CONV_BLK, 256, 0, stream>>>(x, xq_w, xq_b, qb, 0);
  conv1x1_wmma_kernel<8><<<CONV_BLK, 256, 0, stream>>>(x, xk_w, xk_b, kb, 0);
  conv1x1_wmma_kernel<64><<<CONV_BLK, 256, 0, stream>>>(x, xv_w, xv_b, vb, 0);
  cc_softmax_kernel<<<BB * WWID, 128, 0, stream>>>(qb, kb, ah, aw);
  cc_out_h_kernel<<<BB * WWID, 256, 0, stream>>>(vb, ah, oh);
  cc_out_w_combine_kernel<<<BB * HH, 256, 0, stream>>>(vb, aw, oh, x, x_gamma, sx);

  // ---- s_module on y -> sf ----
  conv1x1_wmma_kernel<8><<<CONV_BLK, 256, 0, stream>>>(y, fq_w, fq_b, qb, 0);
  conv1x1_wmma_kernel<8><<<CONV_BLK, 256, 0, stream>>>(y, fk_w, fk_b, kb, 0);
  conv1x1_wmma_kernel<64><<<CONV_BLK, 256, 0, stream>>>(y, fv_w, fv_b, vb, 0);
  cc_softmax_kernel<<<BB * WWID, 128, 0, stream>>>(qb, kb, ah, aw);
  cc_out_h_kernel<<<BB * WWID, 256, 0, stream>>>(vb, ah, oh);
  cc_out_w_combine_kernel<<<BB * HH, 256, 0, stream>>>(vb, aw, oh, y, f_gamma, sf);

  // ---- CAM on (x, y) ----
  cam_energy_kernel<<<BB, 256, 0, stream>>>(y, energy);
  cam_att_kernel<<<BB, 64, 0, stream>>>(energy, attb);
  conv1x1_wmma_kernel<64><<<CONV_BLK, 256, 0, stream>>>(y, attb, nullptr, camv,
                                                        (long)64 * 64);

  // ---- out = pam_out + cam_out = 2x + pg*sx + pg2*sf + cg*(att@f) ----
  final_combine_kernel<<<ELEM_BLK, 256, 0, stream>>>(x, sx, sf, camv,
                                                     pam_g, pam_g2, cam_g, out);
}
